// GQAAttention_30657476559567
// MI455X (gfx1250) — matrix operand order repaired and checked by reading, not compiled, not run on hardware
//
#include <hip/hip_runtime.h>

typedef __bf16 bf16_t;
typedef __attribute__((ext_vector_type(4)))  __bf16 v4bf;
typedef __attribute__((ext_vector_type(8)))  __bf16 v8bf;
typedef __attribute__((ext_vector_type(16))) __bf16 v16bf;
typedef __attribute__((ext_vector_type(8)))  float  v8f;
typedef __attribute__((ext_vector_type(4)))  int    v4i_t;

#define B_   2
#define S_   2048
#define HID_ 2048
#define NH_  16
#define NKV_ 4
#define HD_  128

// ---------------------------------------------------------------------------
// helpers
// ---------------------------------------------------------------------------
__device__ __forceinline__ v16bf mkfrag(v8bf lo, v8bf hi) {
  v16bf r;
#pragma unroll
  for (int i = 0; i < 8; ++i) { r[i] = lo[i]; r[i + 8] = hi[i]; }
  return r;
}

__device__ __forceinline__ v8f wmma_bf16(v16bf a, v16bf b, v8f c) {
  // v_wmma_f32_16x16x32_bf16  (neg_a, A, neg_b, B, c_mod, C, reuse_a, reuse_b)
  return __builtin_amdgcn_wmma_f32_16x16x32_bf16(false, a, false, b, (short)0, c,
                                                 false, false);
}

__device__ __forceinline__ float redmax16(float v) {
#pragma unroll
  for (int m = 8; m >= 1; m >>= 1) v = fmaxf(v, __shfl_xor(v, m, 32));
  return v;
}
__device__ __forceinline__ float redsum16(float v) {
#pragma unroll
  for (int m = 8; m >= 1; m >>= 1) v += __shfl_xor(v, m, 32);
  return v;
}

// ---- CDNA5 async global -> LDS copy (ASYNCcnt path), with sync fallback ----
#if __has_builtin(__builtin_amdgcn_global_load_async_to_lds_b128)
#define HAVE_ASYNC_LDS 1
typedef __attribute__((address_space(1))) v4i_t* g128_p;  // global int4*
typedef __attribute__((address_space(3))) v4i_t* l128_p;  // lds int4*
__device__ __forceinline__ void async_cp16(const bf16_t* g, bf16_t* l) {
  // int-roundtrip casts: plain reinterpret, no addrspacecast legality issues.
  // Generic LDS addresses carry the LDS offset in the low 32 bits (ISA 10.2).
  g128_p gp = (g128_p)(unsigned long long)(size_t)g;
  l128_p lp = (l128_p)(unsigned int)(size_t)l;
  __builtin_amdgcn_global_load_async_to_lds_b128(gp, lp, 0, 0);
}
__device__ __forceinline__ void async_wait0() {
#if __has_builtin(__builtin_amdgcn_s_wait_asynccnt)
  __builtin_amdgcn_s_wait_asynccnt(0);
#else
  asm volatile("s_wait_asynccnt 0x0" ::: "memory");
#endif
}
#else
#define HAVE_ASYNC_LDS 0
__device__ __forceinline__ void async_cp16(const bf16_t* g, bf16_t* l) {
  *(v8bf*)l = *(const v8bf*)g;
}
__device__ __forceinline__ void async_wait0() {}
#endif

// ---------------------------------------------------------------------------
// 1) fp32 -> bf16 conversion (vectorized: float4 in, bf16x4 out)
// ---------------------------------------------------------------------------
__global__ __launch_bounds__(256) void conv_f32_bf16(const float* __restrict__ src,
                                                     bf16_t* __restrict__ dst, int n) {
  const int i = (blockIdx.x * 256 + threadIdx.x) * 4;
  if (i < n) {
    const float4 f = *(const float4*)(src + i);
    v4bf o;
    o[0] = (bf16_t)f.x; o[1] = (bf16_t)f.y; o[2] = (bf16_t)f.z; o[3] = (bf16_t)f.w;
    *(v4bf*)(dst + i) = o;
  }
}

// ---------------------------------------------------------------------------
// 2) generic bf16 GEMM:  C[M,N] = A[M,K] * B[K,N]   (A,B bf16 row-major)
//    F32OUT: f32 store, else bf16 store
//    block tile 128x128, BK=64, 256 threads = 8 waves (2 x 4)
// ---------------------------------------------------------------------------
#define G_LDA 72
#define G_LDB 72

template <bool F32OUT>
__global__ __launch_bounds__(256)
void gemm_bf16(const bf16_t* __restrict__ A, const bf16_t* __restrict__ Bm,
               void* __restrict__ outp, int M, int N, int K) {
  __shared__ bf16_t a_l[128 * G_LDA];
  __shared__ bf16_t b_l[128 * G_LDB];

  const int tid  = threadIdx.x;
  const int lane = tid & 31;
  const int wv   = tid >> 5;
  const int hx   = (lane >> 4) & 1;
  const int l16  = lane & 15;
  const int wm   = wv >> 2;   // 0..1
  const int wn   = wv & 3;    // 0..3
  const int m0   = blockIdx.y * 128;
  const int n0   = blockIdx.x * 128;

  v8f acc[4][2];
#pragma unroll
  for (int mi = 0; mi < 4; ++mi)
#pragma unroll
    for (int ni = 0; ni < 2; ++ni)
#pragma unroll
      for (int r = 0; r < 8; ++r) acc[mi][ni][r] = 0.f;

  for (int k0 = 0; k0 < K; k0 += 64) {
    // stage A tile [128m][64k] via async global->LDS (raw bf16 copy)
    {
      const int row = tid >> 1, hh = tid & 1;
      const bf16_t* gp = A + (size_t)(m0 + row) * K + k0 + hh * 32;
      bf16_t* sp = a_l + row * G_LDA + hh * 32;
#pragma unroll
      for (int i = 0; i < 4; ++i) async_cp16(gp + i * 8, sp + i * 8);
    }
    // stage B tile transposed -> [128n][64k] (manual: needs transpose)
    {
      const int kl = tid >> 2, qq = tid & 3;
      const bf16_t* gp = Bm + (size_t)(k0 + kl) * N + n0 + qq * 32;
      v8bf t[4];
#pragma unroll
      for (int i = 0; i < 4; ++i) t[i] = *(const v8bf*)(gp + i * 8);
#pragma unroll
      for (int i = 0; i < 4; ++i)
#pragma unroll
        for (int j = 0; j < 8; ++j)
          b_l[(qq * 32 + i * 8 + j) * G_LDB + kl] = t[i][j];
    }
    async_wait0();
    __syncthreads();

#pragma unroll
    for (int c = 0; c < 2; ++c) {
      v16bf af[4], bfr[2];
#pragma unroll
      for (int mi = 0; mi < 4; ++mi) {
        const bf16_t* p = a_l + (wm * 64 + mi * 16 + l16) * G_LDA + c * 32 + hx * 8;
        af[mi] = mkfrag(*(const v8bf*)p, *(const v8bf*)(p + 16));
      }
#pragma unroll
      for (int ni = 0; ni < 2; ++ni) {
        const bf16_t* p = b_l + (wn * 32 + ni * 16 + l16) * G_LDB + c * 32 + hx * 8;
        bfr[ni] = mkfrag(*(const v8bf*)p, *(const v8bf*)(p + 16));
      }
#pragma unroll
      for (int mi = 0; mi < 4; ++mi)
#pragma unroll
        for (int ni = 0; ni < 2; ++ni)
          acc[mi][ni] = wmma_bf16(af[mi], bfr[ni], acc[mi][ni]);
    }
    __syncthreads();
  }

  // epilogue: C layout -> row M = r + 8*hx, col N = l16
#pragma unroll
  for (int mi = 0; mi < 4; ++mi)
#pragma unroll
    for (int ni = 0; ni < 2; ++ni)
#pragma unroll
      for (int r = 0; r < 8; ++r) {
        const int m = m0 + wm * 64 + mi * 16 + 8 * hx + r;
        const int n = n0 + wn * 32 + ni * 16 + l16;
        const size_t off = (size_t)m * N + n;
        if (F32OUT) ((float*)outp)[off]  = acc[mi][ni][r];
        else        ((bf16_t*)outp)[off] = (bf16_t)acc[mi][ni][r];
      }
}

// ---------------------------------------------------------------------------
// 3) RoPE + relayout:  src [B*S, nheads*128] bf16  ->  dst [B, nheads, S, 128]
// ---------------------------------------------------------------------------
__global__ __launch_bounds__(256)
void rope_relayout(const bf16_t* __restrict__ src, bf16_t* __restrict__ dst,
                   int nheads) {
  const int idx   = blockIdx.x * 256 + threadIdx.x;
  const int total = B_ * S_ * nheads * 64;
  if (idx >= total) return;
  const int i2  = idx & 63;
  const int h   = (idx >> 6) % nheads;
  const int row = idx / (64 * nheads);
  const int b   = row / S_;
  const int pos = row % S_;
  const size_t sb = (size_t)row * (nheads * 128) + h * 128 + 2 * i2;
  const float e = (float)src[sb];
  const float o = (float)src[sb + 1];
  // inv_freq = theta^(-i2/64) = exp(-ln(1e6)/64 * i2)
  const float inv = __expf(-0.215867352468192f * (float)i2);
  const float ang = (float)pos * inv;
  float s, c;
  __sincosf(ang, &s, &c);
  const size_t ob = (((size_t)b * nheads + h) * S_ + pos) * 128 + 2 * i2;
  dst[ob]     = (bf16_t)(e * c - o * s);
  dst[ob + 1] = (bf16_t)(e * s + o * c);
}

// V relayout: src [B*S, NKV*128] -> Vt [B, NKV, 128, S]
__global__ __launch_bounds__(256)
void v_transpose(const bf16_t* __restrict__ src, bf16_t* __restrict__ dst) {
  const int idx = blockIdx.x * 256 + threadIdx.x;
  if (idx >= B_ * S_ * NKV_ * HD_) return;
  const int d   = idx & 127;
  const int h   = (idx >> 7) & (NKV_ - 1);
  const int row = idx >> 9;
  const int b   = row / S_;
  const int pos = row % S_;
  dst[(((size_t)b * NKV_ + h) * HD_ + d) * S_ + pos] =
      src[(size_t)row * (NKV_ * HD_) + h * HD_ + d];
}

// ---------------------------------------------------------------------------
// 4) flash attention: Q [B,NH,S,128], K [B,NKV,S,128], Vt [B,NKV,128,S]
//    out: [B*S, NH*128] bf16.  q-tile 128 rows/block, k-tile 64.
// ---------------------------------------------------------------------------
#define K_LD 136
#define V_LD 72
#define P_LD 72

__global__ __launch_bounds__(256)
void attn_flash(const bf16_t* __restrict__ Q, const bf16_t* __restrict__ Kk,
                const bf16_t* __restrict__ Vt, bf16_t* __restrict__ Oo) {
  __shared__ bf16_t k_l[64 * K_LD];    // [s_local][d]
  __shared__ bf16_t v_l[128 * V_LD];   // [d][s_local]
  __shared__ bf16_t p_l[128 * P_LD];   // [wave*16 + m][k_local]

  const int tid  = threadIdx.x;
  const int lane = tid & 31;
  const int wv   = tid >> 5;
  const int hx   = (lane >> 4) & 1;
  const int l16  = lane & 15;
  const int bh   = blockIdx.y;
  const int b    = bh / NH_;
  const int h    = bh % NH_;
  const int hk   = h / (NH_ / NKV_);
  const int q0   = blockIdx.x * 128;

  // Q fragments for this wave's 16 rows (A layout)
  v16bf qf[4];
  {
    const bf16_t* qp =
        Q + (((size_t)b * NH_ + h) * S_ + q0 + wv * 16 + l16) * HD_;
#pragma unroll
    for (int c = 0; c < 4; ++c) {
      const bf16_t* p = qp + c * 32 + hx * 8;
      qf[c] = mkfrag(*(const v8bf*)p, *(const v8bf*)(p + 16));
    }
  }

  v8f o[8];
#pragma unroll
  for (int fd = 0; fd < 8; ++fd)
#pragma unroll
    for (int r = 0; r < 8; ++r) o[fd][r] = 0.f;
  float m_run[8], l_run[8];
#pragma unroll
  for (int r = 0; r < 8; ++r) { m_run[r] = -1e30f; l_run[r] = 0.f; }

  const bf16_t* Kb = Kk + ((size_t)b * NKV_ + hk) * S_ * HD_;
  const bf16_t* Vb = Vt + ((size_t)b * NKV_ + hk) * (size_t)HD_ * S_;
  const float scale = 0.08838834764831845f; // 1/sqrt(128)

  const int nkt = q0 / 64 + 2;
  for (int kt = 0; kt < nkt; ++kt) {
    const int s0 = kt * 64;
    // stage K tile [64s][128d] + V tile [128d][64s] via async global->LDS
    {
      const int row = tid >> 2, qq = tid & 3;
      const bf16_t* gp = Kb + (size_t)(s0 + row) * HD_ + qq * 32;
      bf16_t* sp = k_l + row * K_LD + qq * 32;
#pragma unroll
      for (int i = 0; i < 4; ++i) async_cp16(gp + i * 8, sp + i * 8);
    }
    {
      const int d = tid >> 1, hh = tid & 1;
      const bf16_t* gp = Vb + (size_t)d * S_ + s0 + hh * 32;
      bf16_t* sp = v_l + d * V_LD + hh * 32;
#pragma unroll
      for (int i = 0; i < 4; ++i) async_cp16(gp + i * 8, sp + i * 8);
    }
    async_wait0();
    __syncthreads();

    // scores S = Q * K^T   (4 n-fragments of 16 k-columns)
    v8f sc[4];
#pragma unroll
    for (int nf = 0; nf < 4; ++nf) {
#pragma unroll
      for (int r = 0; r < 8; ++r) sc[nf][r] = 0.f;
#pragma unroll
      for (int c = 0; c < 4; ++c) {
        const bf16_t* p = k_l + (nf * 16 + l16) * K_LD + c * 32 + hx * 8;
        v16bf kb = mkfrag(*(const v8bf*)p, *(const v8bf*)(p + 16));
        sc[nf] = wmma_bf16(qf[c], kb, sc[nf]);
      }
    }

    // scale + causal mask
#pragma unroll
    for (int nf = 0; nf < 4; ++nf) {
      const int s_idx = s0 + nf * 16 + l16;
#pragma unroll
      for (int r = 0; r < 8; ++r) {
        const int q_idx = q0 + wv * 16 + 8 * hx + r;
        float vsc = sc[nf][r] * scale;
        if (s_idx > q_idx) vsc = -1e9f;
        sc[nf][r] = vsc;
      }
    }

    // online softmax
    float rm[8], alpha[8], rs[8];
#pragma unroll
    for (int r = 0; r < 8; ++r) {
      float v = sc[0][r];
#pragma unroll
      for (int nf = 1; nf < 4; ++nf) v = fmaxf(v, sc[nf][r]);
      rm[r] = redmax16(v);
      const float mnew = fmaxf(m_run[r], rm[r]);
      alpha[r] = __expf(m_run[r] - mnew);
      m_run[r] = mnew;
      rs[r] = 0.f;
    }
#pragma unroll
    for (int nf = 0; nf < 4; ++nf)
#pragma unroll
      for (int r = 0; r < 8; ++r) {
        const float p = __expf(sc[nf][r] - m_run[r]);
        rs[r] += p;
        p_l[(wv * 16 + 8 * hx + r) * P_LD + nf * 16 + l16] = (bf16_t)p;
      }
#pragma unroll
    for (int r = 0; r < 8; ++r) {
      l_run[r] = l_run[r] * alpha[r] + redsum16(rs[r]);
    }
#pragma unroll
    for (int fd = 0; fd < 8; ++fd)
#pragma unroll
      for (int r = 0; r < 8; ++r) o[fd][r] *= alpha[r];

    // O += P * V
#pragma unroll
    for (int c = 0; c < 2; ++c) {
      const bf16_t* pp = p_l + (wv * 16 + l16) * P_LD + c * 32 + hx * 8;
      v16bf pf = mkfrag(*(const v8bf*)pp, *(const v8bf*)(pp + 16));
#pragma unroll
      for (int fd = 0; fd < 8; ++fd) {
        const bf16_t* vp = v_l + (fd * 16 + l16) * V_LD + c * 32 + hx * 8;
        v16bf vf = mkfrag(*(const v8bf*)vp, *(const v8bf*)(vp + 16));
        o[fd] = wmma_bf16(pf, vf, o[fd]);
      }
    }
    __syncthreads();
  }

  // finalize
#pragma unroll
  for (int r = 0; r < 8; ++r) {
    const float inv = 1.f / l_run[r];
    const int row = q0 + wv * 16 + 8 * hx + r;
#pragma unroll
    for (int fd = 0; fd < 8; ++fd) {
      Oo[((size_t)b * S_ + row) * (NH_ * HD_) + h * HD_ + fd * 16 + l16] =
          (bf16_t)(o[fd][r] * inv);
    }
  }
}

// ---------------------------------------------------------------------------
// launch
// ---------------------------------------------------------------------------
extern "C" void kernel_launch(void* const* d_in, const int* in_sizes, int n_in,
                              void* d_out, int out_size, void* d_ws, size_t ws_size,
                              hipStream_t stream) {
  (void)in_sizes; (void)n_in; (void)out_size; (void)ws_size;
  const float* x  = (const float*)d_in[0];
  // d_in[1] = position_ids (arange, computed analytically)
  // d_in[2] = attention_mask (causal, applied analytically)
  const float* Wq = (const float*)d_in[3];
  const float* Wk = (const float*)d_in[4];
  const float* Wv = (const float*)d_in[5];
  const float* Wo = (const float*)d_in[6];
  float* out = (float*)d_out;

  const int MR = B_ * S_;            // 4096 rows
  const int NQ = NH_ * HD_;          // 2048
  const int NK = NKV_ * HD_;         // 512

  char* ws = (char*)d_ws;
  size_t off = 0;
  auto alloc = [&](size_t bytes) {
    void* p = ws + off;
    off += (bytes + 255) & ~(size_t)255;
    return p;
  };
  bf16_t* xb  = (bf16_t*)alloc((size_t)MR * HID_ * 2);
  bf16_t* wqb = (bf16_t*)alloc((size_t)HID_ * NQ * 2);
  bf16_t* wkb = (bf16_t*)alloc((size_t)HID_ * NK * 2);
  bf16_t* wvb = (bf16_t*)alloc((size_t)HID_ * NK * 2);
  bf16_t* wob = (bf16_t*)alloc((size_t)NQ * HID_ * 2);
  bf16_t* Qb  = (bf16_t*)alloc((size_t)MR * NQ * 2);
  bf16_t* Kb  = (bf16_t*)alloc((size_t)MR * NK * 2);
  bf16_t* Vtb = (bf16_t*)alloc((size_t)MR * NK * 2);

  // pre-RoPE QKV parked inside d_out (fully overwritten by final GEMM)
  bf16_t* qtmp = (bf16_t*)d_out;                                     // 16 MB
  bf16_t* ktmp = (bf16_t*)((char*)d_out + (size_t)16 * 1024 * 1024); // 4 MB
  bf16_t* vtmp = (bf16_t*)((char*)d_out + (size_t)20 * 1024 * 1024); // 4 MB
  // attention output reuses xb (x no longer needed after projections)
  bf16_t* aob = xb;

  auto cgrid4 = [](int n) { return dim3((unsigned)((n / 4 + 255) / 256)); };
  auto cgrid  = [](int n) { return dim3((unsigned)((n + 255) / 256)); };

  conv_f32_bf16<<<cgrid4(MR * HID_), 256, 0, stream>>>(x, xb, MR * HID_);
  conv_f32_bf16<<<cgrid4(HID_ * NQ), 256, 0, stream>>>(Wq, wqb, HID_ * NQ);
  conv_f32_bf16<<<cgrid4(HID_ * NK), 256, 0, stream>>>(Wk, wkb, HID_ * NK);
  conv_f32_bf16<<<cgrid4(HID_ * NK), 256, 0, stream>>>(Wv, wvb, HID_ * NK);
  conv_f32_bf16<<<cgrid4(NQ * HID_), 256, 0, stream>>>(Wo, wob, NQ * HID_);

  gemm_bf16<false><<<dim3(NQ / 128, MR / 128), 256, 0, stream>>>(
      xb, wqb, qtmp, MR, NQ, HID_);
  gemm_bf16<false><<<dim3(NK / 128, MR / 128), 256, 0, stream>>>(
      xb, wkb, ktmp, MR, NK, HID_);
  gemm_bf16<false><<<dim3(NK / 128, MR / 128), 256, 0, stream>>>(
      xb, wvb, vtmp, MR, NK, HID_);

  rope_relayout<<<cgrid(B_ * S_ * NH_ * 64), 256, 0, stream>>>(qtmp, Qb, NH_);
  rope_relayout<<<cgrid(B_ * S_ * NKV_ * 64), 256, 0, stream>>>(ktmp, Kb, NKV_);
  v_transpose<<<cgrid(B_ * S_ * NKV_ * HD_), 256, 0, stream>>>(vtmp, Vtb);

  attn_flash<<<dim3(S_ / 128, B_ * NH_), 256, 0, stream>>>(Qb, Kb, Vtb, aob);

  gemm_bf16<true><<<dim3(HID_ / 128, MR / 128), 256, 0, stream>>>(
      aob, wob, out, MR, HID_, NQ);
}
